// MultiQueryAttention_13030930776442
// MI455X (gfx1250) — compile-verified, matrix operand order checked by reading, not run on hardware
//
#include <hip/hip_runtime.h>
#include <hip/hip_bf16.h>
#include <math.h>

// ---- problem constants (match reference) ----
#define DIMC 1024
#define NH   16
#define NKV  4
#define HD   64
#define BB   2
#define TT   2048
#define BT   (BB * TT)     // 4096 rows
#define KVD  (NKV * HD)    // 256

typedef __attribute__((ext_vector_type(16))) __bf16 v16bf;
typedef __attribute__((ext_vector_type(8)))  __bf16 v8bf;
typedef __attribute__((ext_vector_type(4)))  __bf16 v4bf;
typedef __attribute__((ext_vector_type(8)))  float  v8f;
typedef __attribute__((ext_vector_type(4)))  float  v4f;
typedef __attribute__((ext_vector_type(4)))  int    v4i;
typedef __attribute__((ext_vector_type(8)))  int    v8i;
typedef __attribute__((ext_vector_type(4)))  unsigned int v4u;

// pointer-to-vector types in the address spaces the async builtin expects
typedef __attribute__((address_space(1))) v4i* gv4i_p;   // global
typedef __attribute__((address_space(3))) v4i* lv4i_p;   // LDS

#if defined(__has_builtin)
#if __has_builtin(__builtin_amdgcn_global_load_async_to_lds_b128) && \
    __has_builtin(__builtin_amdgcn_s_wait_asynccnt)
#define HAVE_ASYNC_LDS 1
#endif
#if __has_builtin(__builtin_amdgcn_tensor_load_to_lds) && \
    __has_builtin(__builtin_amdgcn_s_wait_tensorcnt)
#define HAVE_TDM 1
#endif
#endif

// generic pointer (to a __shared__ object) -> 32-bit LDS byte offset
__device__ __forceinline__ unsigned lds_offset(const void* p) {
  return (unsigned)(unsigned long long)(uintptr_t)
      (__attribute__((address_space(3))) const void*)p;
}

#if defined(HAVE_TDM)
// TDM: DMA one contiguous 4KB (2048 x 2B) chunk from global memory into LDS.
// D# group0: count=1 | lds_addr | global_addr[56:0] | type=2
// D# group1: data_size=2B; tensor_dim0 = tile_dim0 = 2048 (1-D tile)
// This toolchain's builtin is the 6-arg form:
//   (uint32x4 g0, int32x8 g1, int32x4 g2, int32x4 g3, int32x8 pad, i32 cpol)
__device__ __forceinline__ void tdm_copy_4kb(const void* gsrc, unsigned lds_off) {
  unsigned long long ga = (unsigned long long)(uintptr_t)gsrc;
  v4u g0;
  g0[0] = 1u;                                            // count=1 (valid D#)
  g0[1] = lds_off;                                       // lds_addr (bytes)
  g0[2] = (unsigned)(ga & 0xFFFFFFFFu);                  // global_addr[31:0]
  g0[3] = (unsigned)((ga >> 32) & 0x01FFFFFFu)           // global_addr[56:32]
          | 0x80000000u;                                 // type=2 ("image")
  v8i g1;
  g1[0] = 0x00010000;               // data_size=1 (2 bytes); no multicast/pad
  g1[1] = (int)(2048u << 16);       // tensor_dim0[15:0]=2048 (bits 63:48)
  g1[2] = 0;                        // tensor_dim0[31:16]=0, tensor_dim1 lo=0
  g1[3] = (int)(2048u << 16);       // tile_dim0=2048 (bits 127:112)
  g1[4] = 0;                        // tile_dim1=0 (unused), tile_dim2=0
  g1[5] = 2048;                     // tensor_dim0_stride lo (unused for 1-D)
  g1[6] = 0;
  g1[7] = 0;
  v4i zero4 = {0, 0, 0, 0};
  v8i zero8 = {0, 0, 0, 0, 0, 0, 0, 0};
  __builtin_amdgcn_tensor_load_to_lds(g0, g1, zero4, zero4, zero8, 0);
}
#endif

// Build a 16-element bf16 fragment from two contiguous 8-element chunks.
// Per ISA 7.12.2 (16-bit A/B layout): lane half selects khalf (0/8); elements
// 0..7 cover K = khalf..khalf+7, elements 8..15 cover K = 16+khalf..16+khalf+7.
__device__ __forceinline__ v16bf make_frag(const __bf16* lo, const __bf16* hi) {
  v8bf a = *(const v8bf*)lo;
  v8bf b = *(const v8bf*)hi;
  v16bf r;
#pragma unroll
  for (int e = 0; e < 8; ++e) { r[e] = a[e]; r[e + 8] = b[e]; }
  return r;
}

// ---------------------------------------------------------------------------
// Generic GEMM: C[M,N] = A[M,K] @ W[K,N] + bias[N]   (fp32 in/out, bf16 WMMA)
// Block: 256 threads (8 waves). Tile: 128 (M) x 128 (N), K-step 32.
// Wave w computes rows [w*16, w*16+16) of the tile, 8 column fragments.
// ---------------------------------------------------------------------------
__global__ __launch_bounds__(256) void gemm_bias_wmma(
    const float* __restrict__ A, const float* __restrict__ W,
    const float* __restrict__ bias, float* __restrict__ C,
    int M, int N, int K) {
  __shared__ __bf16 As[128][32];    // [m][k]  8 KB
  __shared__ __bf16 Bs[128][32];    // [n][k]  8 KB (W tile stored col-major)
  const int t = threadIdx.x;
  const int lane = t & 31, wave = t >> 5;
  const int lrow = lane & 15;
  const int khalf = (lane >> 4) << 3;   // 0 or 8
  const int hi8 = khalf;                // C/D: lanes 16-31 hold rows +8
  const int m0 = blockIdx.y * 128;
  const int n0 = blockIdx.x * 128;

  v8f acc[8];
#pragma unroll
  for (int nb = 0; nb < 8; ++nb)
#pragma unroll
    for (int e = 0; e < 8; ++e) acc[nb][e] = 0.f;

  for (int k0 = 0; k0 < K; k0 += 32) {
    // A tile 128x32: 4 x float4 per thread, vector cvt + b64 LDS store
#pragma unroll
    for (int i = 0; i < 4; ++i) {
      int base = (i * 256 + t) * 4;
      int r = base >> 5, c = base & 31;
      v4f f = *(const v4f*)&A[(size_t)(m0 + r) * K + k0 + c];
      v4bf hbits;
#pragma unroll
      for (int j = 0; j < 4; ++j) hbits[j] = (__bf16)f[j];
      *(v4bf*)&As[r][c] = hbits;
    }
    // B tile 32x128: 4 x float4 per thread, transposed scalar stores
#pragma unroll
    for (int i = 0; i < 4; ++i) {
      int base = (i * 256 + t) * 4;
      int kk = base >> 7, n = base & 127;
      v4f f = *(const v4f*)&W[(size_t)(k0 + kk) * N + n0 + n];
#pragma unroll
      for (int j = 0; j < 4; ++j) Bs[n + j][kk] = (__bf16)f[j];
    }
    __syncthreads();
    v16bf af = make_frag(&As[wave * 16 + lrow][khalf],
                         &As[wave * 16 + lrow][16 + khalf]);
#pragma unroll
    for (int nb = 0; nb < 8; ++nb) {
      v16bf bf = make_frag(&Bs[nb * 16 + lrow][khalf],
                           &Bs[nb * 16 + lrow][16 + khalf]);
      acc[nb] = __builtin_amdgcn_wmma_f32_16x16x32_bf16(
          false, af, false, bf, (short)0, acc[nb], false, false);
    }
    __syncthreads();
  }

#pragma unroll
  for (int nb = 0; nb < 8; ++nb) {
    int n = n0 + nb * 16 + lrow;
    float bv = bias[n];
#pragma unroll
    for (int r = 0; r < 8; ++r) {
      int m = m0 + wave * 16 + r + hi8;
      C[(size_t)m * N + n] = acc[nb][r] + bv;
    }
  }
}

// ---------------------------------------------------------------------------
// RoPE (reference semantics) + transpose to [B,H,T,HD], fp32 -> bf16.
// out[j]      = x[2j]*cos - x[2j+1]*sin
// out[j+HD/2] = x[2j]*sin - x[2j+1]*cos      (as written in the reference)
// ---------------------------------------------------------------------------
__global__ void rope_t_kernel(const float* __restrict__ src, __bf16* __restrict__ dst,
                              int H, int total) {
  int idx = blockIdx.x * blockDim.x + threadIdx.x;
  if (idx >= total) return;
  int j = idx & 31;
  int tt = (idx >> 5) & (TT - 1);
  int h = (idx >> 16) % H;              // 32 * 2048 = 65536 per (b,h)
  int b = idx / (32 * TT * H);
  const float* p = src + ((size_t)(b * TT + tt) * (H * HD)) + h * HD;
  float x1 = p[2 * j], x2 = p[2 * j + 1];
  float f = (float)tt * __powf(10000.f, -(float)(2 * j) / (float)HD);
  float s, c;
  __sincosf(f, &s, &c);
  __bf16* q = dst + ((size_t)((b * H + h) * TT + tt) * HD);
  q[j]      = (__bf16)(x1 * c - x2 * s);
  q[j + 32] = (__bf16)(x1 * s - x2 * c);
}

// V transpose [B,T,H*HD] fp32 -> blocked-transposed bf16 [B,H,T/32, HD, 32]
// so a 32-key V block is one contiguous 4KB chunk already in [hd][key] order.
__global__ void copy_t_v_kernel(const float* __restrict__ src, __bf16* __restrict__ dst,
                                int H, int total) {
  int idx = blockIdx.x * blockDim.x + threadIdx.x;
  if (idx >= total) return;
  int d = idx & 63;
  int tt = (idx >> 6) & (TT - 1);
  int h = (idx >> 17) % H;              // 64 * 2048 = 131072 per (b,h)
  int b = idx / (64 * TT * H);
  int blk = tt >> 5, kib = tt & 31;
  dst[(size_t)((b * H + h) * (TT / 32) + blk) * (HD * 32) + d * 32 + kib] =
      (__bf16)src[(size_t)(b * TT + tt) * (H * HD) + h * HD + d];
}

// ---------------------------------------------------------------------------
// Causal flash attention, bf16 WMMA, fp32 online softmax.
// Grid: (T/128, NH, B). Block: 256 (8 waves); wave w owns q rows [w*16,w*16+16).
// 32-key K/V blocks staged in LDS via the Tensor Data Mover (TENSORcnt) when
// available, else GLOBAL_LOAD_ASYNC_TO_LDS_B128 (ASYNCcnt). P round-trips
// through per-wave LDS to convert the WMMA D-fragment layout into A-layout.
// ---------------------------------------------------------------------------
__global__ __launch_bounds__(256) void flash_attn_wmma(
    const __bf16* __restrict__ Qr, const __bf16* __restrict__ Kr,
    const __bf16* __restrict__ Vr, float* __restrict__ Ctx) {
  __shared__ __bf16 Ks[32][64];        // [key][hd]   4 KB
  __shared__ __bf16 Vst[64][32];       // [hd][key]   4 KB (arrives pre-transposed)
  __shared__ __bf16 Ps[8][16][32];     // per-wave P  8 KB
  const int t = threadIdx.x;
  const int lane = t & 31, wave = t >> 5;
  const int lrow = lane & 15;
  const int khalf = (lane >> 4) << 3;
  const int hi8 = khalf;
  const int q0 = blockIdx.x * 128;
  const int h = blockIdx.y;
  const int b = blockIdx.z;
  const int kvh = h / (NH / NKV);

  const __bf16* Qrow = Qr + ((size_t)((b * NH + h) * TT) + q0 + wave * 16 + lrow) * HD;
  const __bf16* Kbase = Kr + (size_t)((b * NKV + kvh) * TT) * HD;
  const __bf16* Vblk0 = Vr + (size_t)((b * NKV + kvh) * (TT / 32)) * (HD * 32);

  // Q fragments (K-dim = HD = 64 -> two 16x16x32 steps)
  v16bf qf0 = make_frag(Qrow + khalf, Qrow + 16 + khalf);
  v16bf qf1 = make_frag(Qrow + 32 + khalf, Qrow + 48 + khalf);

  v8f o[4];
  float mrow[8], lsum[8];
#pragma unroll
  for (int nb = 0; nb < 4; ++nb)
#pragma unroll
    for (int e = 0; e < 8; ++e) o[nb][e] = 0.f;
#pragma unroll
  for (int r = 0; r < 8; ++r) { mrow[r] = -1e30f; lsum[r] = 0.f; }

  const int qend = q0 + wave * 16 + 15;
  const int nkb = (q0 >> 5) + 4;       // key blocks needed for this q tile
  const float scale = 0.125f;          // 1/sqrt(64)

  for (int kb = 0; kb < nkb; ++kb) {
    const int kbase = kb * 32;
    // ---- stage K (contiguous rows) and V (pre-transposed block) into LDS ----
    {
#if defined(HAVE_TDM)
      if (wave == 0) {
        tdm_copy_4kb(Kbase + (size_t)kbase * HD, lds_offset(&Ks[0][0]));
        tdm_copy_4kb(Vblk0 + (size_t)kb * (HD * 32), lds_offset(&Vst[0][0]));
      }
      __builtin_amdgcn_s_wait_tensorcnt(0);
#elif defined(HAVE_ASYNC_LDS)
      const __bf16* ksrc = Kbase + (size_t)kbase * HD + t * 8;
      const __bf16* vsrc = Vblk0 + (size_t)kb * (HD * 32) + t * 8;
      __builtin_amdgcn_global_load_async_to_lds_b128(
          (gv4i_p)(const void*)ksrc, (lv4i_p)(void*)(&Ks[0][0] + t * 8), 0, 0);
      __builtin_amdgcn_global_load_async_to_lds_b128(
          (gv4i_p)(const void*)vsrc, (lv4i_p)(void*)(&Vst[0][0] + t * 8), 0, 0);
      __builtin_amdgcn_s_wait_asynccnt(0);
#else
      *(v8bf*)(&Ks[0][0] + t * 8)  = *(const v8bf*)(Kbase + (size_t)kbase * HD + t * 8);
      *(v8bf*)(&Vst[0][0] + t * 8) = *(const v8bf*)(Vblk0 + (size_t)kb * (HD * 32) + t * 8);
#endif
      if (kb + 1 < nkb) {  // prefetch next block while we compute this one
        __builtin_prefetch(Kbase + (size_t)(kbase + 32) * HD + t * 8, 0, 0);
        __builtin_prefetch(Vblk0 + (size_t)(kb + 1) * (HD * 32) + t * 8, 0, 0);
      }
    }
    __syncthreads();

    if (kbase <= qend) {               // wave has at least one unmasked key
      v8f s[2];
#pragma unroll
      for (int sub = 0; sub < 2; ++sub) {
#pragma unroll
        for (int e = 0; e < 8; ++e) s[sub][e] = 0.f;
        const __bf16* krow = &Ks[sub * 16 + lrow][0];
        v16bf kf0 = make_frag(krow + khalf, krow + 16 + khalf);
        v16bf kf1 = make_frag(krow + 32 + khalf, krow + 48 + khalf);
        s[sub] = __builtin_amdgcn_wmma_f32_16x16x32_bf16(
            false, qf0, false, kf0, (short)0, s[sub], false, false);
        s[sub] = __builtin_amdgcn_wmma_f32_16x16x32_bf16(
            false, qf1, false, kf1, (short)0, s[sub], false, false);
      }
      // scale + causal mask
      const int qrow_base = q0 + wave * 16 + hi8;
#pragma unroll
      for (int sub = 0; sub < 2; ++sub) {
        int kg = kbase + sub * 16 + lrow;
#pragma unroll
        for (int r = 0; r < 8; ++r) {
          float v = s[sub][r] * scale;
          s[sub][r] = (kg <= qrow_base + r) ? v : -1e30f;
        }
      }
      // online softmax (row lives across a 16-lane half within each VGPR)
      float alpha[8], mnew[8];
#pragma unroll
      for (int r = 0; r < 8; ++r) {
        float mx = fmaxf(s[0][r], s[1][r]);
#pragma unroll
        for (int off = 1; off < 16; off <<= 1)
          mx = fmaxf(mx, __shfl_xor(mx, off, 32));
        mnew[r] = fmaxf(mrow[r], mx);
        alpha[r] = __expf(mrow[r] - mnew[r]);
        mrow[r] = mnew[r];
      }
#pragma unroll
      for (int r = 0; r < 8; ++r) {
        float p0 = __expf(s[0][r] - mnew[r]);
        float p1 = __expf(s[1][r] - mnew[r]);
        float rs = p0 + p1;
#pragma unroll
        for (int off = 1; off < 16; off <<= 1) rs += __shfl_xor(rs, off, 32);
        lsum[r] = lsum[r] * alpha[r] + rs;
        Ps[wave][r + hi8][lrow] = (__bf16)p0;        // D layout -> LDS
        Ps[wave][r + hi8][16 + lrow] = (__bf16)p1;
      }
#pragma unroll
      for (int nb = 0; nb < 4; ++nb)
#pragma unroll
        for (int r = 0; r < 8; ++r) o[nb][r] *= alpha[r];

      __builtin_amdgcn_wave_barrier();               // order per-wave LDS st->ld
      // P as A-fragment (16 q-rows x 32 keys), V columns as B-fragments
      v16bf pf = make_frag(&Ps[wave][lrow][khalf], &Ps[wave][lrow][16 + khalf]);
#pragma unroll
      for (int nb = 0; nb < 4; ++nb) {
        const __bf16* vrow = &Vst[nb * 16 + lrow][0];
        v16bf vf = make_frag(vrow + khalf, vrow + 16 + khalf);
        o[nb] = __builtin_amdgcn_wmma_f32_16x16x32_bf16(
            false, pf, false, vf, (short)0, o[nb], false, false);
      }
    }
    __syncthreads();
  }

  // normalize and store ctx in [B,T,NH*HD] (GEMM-ready)
#pragma unroll
  for (int r = 0; r < 8; ++r) {
    float inv = 1.f / lsum[r];
    int tq = q0 + wave * 16 + r + hi8;
#pragma unroll
    for (int nb = 0; nb < 4; ++nb)
      Ctx[(size_t)(b * TT + tq) * DIMC + h * HD + nb * 16 + lrow] = o[nb][r] * inv;
  }
}

// ---------------------------------------------------------------------------
extern "C" void kernel_launch(void* const* d_in, const int* in_sizes, int n_in,
                              void* d_out, int out_size, void* d_ws, size_t ws_size,
                              hipStream_t stream) {
  const float* x  = (const float*)d_in[0];
  const float* Wq = (const float*)d_in[1];
  const float* bq = (const float*)d_in[2];
  const float* Wk = (const float*)d_in[3];
  const float* bk = (const float*)d_in[4];
  const float* Wv = (const float*)d_in[5];
  const float* bv = (const float*)d_in[6];
  const float* Wo = (const float*)d_in[7];
  const float* bo = (const float*)d_in[8];
  float* out = (float*)d_out;

  char* ws = (char*)d_ws;
  size_t off = 0;
  float*  Qlin = (float*)(ws + off); off += (size_t)BT * DIMC * sizeof(float);
  float*  Klin = (float*)(ws + off); off += (size_t)BT * KVD * sizeof(float);
  float*  Vlin = (float*)(ws + off); off += (size_t)BT * KVD * sizeof(float);
  __bf16* Qr   = (__bf16*)(ws + off); off += (size_t)BT * DIMC * sizeof(__bf16);
  __bf16* Kr   = (__bf16*)(ws + off); off += (size_t)BT * KVD * sizeof(__bf16);
  __bf16* Vr   = (__bf16*)(ws + off); off += (size_t)BT * KVD * sizeof(__bf16);
  float*  Ctx  = (float*)(ws + off); off += (size_t)BT * DIMC * sizeof(float);

  dim3 blk(256);
  // QKV projections
  gemm_bias_wmma<<<dim3(DIMC / 128, BT / 128), blk, 0, stream>>>(x, Wq, bq, Qlin, BT, DIMC, DIMC);
  gemm_bias_wmma<<<dim3(KVD / 128, BT / 128), blk, 0, stream>>>(x, Wk, bk, Klin, BT, KVD, DIMC);
  gemm_bias_wmma<<<dim3(KVD / 128, BT / 128), blk, 0, stream>>>(x, Wv, bv, Vlin, BT, KVD, DIMC);
  // RoPE + transpose to head-major bf16
  int nq = BB * NH * TT * (HD / 2);
  rope_t_kernel<<<(nq + 255) / 256, blk, 0, stream>>>(Qlin, Qr, NH, nq);
  int nk = BB * NKV * TT * (HD / 2);
  rope_t_kernel<<<(nk + 255) / 256, blk, 0, stream>>>(Klin, Kr, NKV, nk);
  int nv = BB * NKV * TT * HD;
  copy_t_v_kernel<<<(nv + 255) / 256, blk, 0, stream>>>(Vlin, Vr, NKV, nv);
  // causal flash attention
  flash_attn_wmma<<<dim3(TT / 128, NH, BB), blk, 0, stream>>>(Qr, Kr, Vr, Ctx);
  // output projection
  gemm_bias_wmma<<<dim3(DIMC / 128, BT / 128), blk, 0, stream>>>(Ctx, Wo, bo, out, BT, DIMC, DIMC);
}
